// GAT_89215060673335
// MI455X (gfx1250) — matrix operand order repaired and checked by reading, not compiled, not run on hardware
//
#include <hip/hip_runtime.h>
#include <hip/hip_bf16.h>
#include <math.h>

typedef __bf16 v16bf __attribute__((ext_vector_type(16)));
typedef __bf16 bf8v  __attribute__((ext_vector_type(8)));
typedef float  v8f   __attribute__((ext_vector_type(8)));

#define NEG_SLOPE 0.2f
#define BN_EPS    1e-5f

__device__ __forceinline__ float lrelu(float x) { return x > 0.f ? x : NEG_SLOPE * x; }

__device__ __forceinline__ void atomicMaxF(float* addr, float val) {
  int* ai = (int*)addr;
  int old = __float_as_int(*addr);
  while (__int_as_float(old) < val) {
    int assumed = old;
    old = atomicCAS(ai, assumed, __float_as_int(val));
    if (old == assumed) break;
  }
}

// ---------------------------------------------------------------------------
// Generic fill
// ---------------------------------------------------------------------------
__global__ void k_fill(float* __restrict__ p, float v, long long n) {
  long long i = (long long)blockIdx.x * blockDim.x + threadIdx.x;
  if (i < n) p[i] = v;
}

// ---------------------------------------------------------------------------
// WMMA GEMM with LDS staging: C[nrows, FOUT] = A[nrows, 128] * B[128, FOUT]
// Block = 256 threads = 8 waves:
//   1) stage 64 rows of A (fp32 -> bf16) into LDS  [row][k], padded rows
//   2) stage all of B transposed (bf16) into LDS   [col][k], padded rows
//   3) 8 waves sweep 16x16 output tiles; A/B fragments are 16B LDS vector
//      loads (conflict-free via +8-element row padding). B fragments are
//      loop-invariant per wave (colTile fixed) and stay in registers.
// All 8 A-chunks for the 4 k-steps are preloaded before the 4 WMMAs so the
// compiler can overlap ds_loads with WMMA issue.
// C stores: block-uniform fullBlock branch -> clause of 8 immediate-offset
// global_store_b32 (guarded path only for the single partial block).
// Fragment layouts per CDNA5 ISA 7.12.2 (wave32):
//   A (16x32): lane<16:  M=lane,    i<8 -> K=i,    i>=8 -> K=i+8
//              lane>=16: M=lane-16, i<8 -> K=i+8,  i>=8 -> K=i+16
//   B (32x16): N=lane%16, K=(lane>=16?8:0)+i+(i>=8?8:0), as for A
//   C/D f32 :  VGPR r -> M = r + (lane>=16 ? 8 : 0), N = lane%16
// ---------------------------------------------------------------------------
template<int FOUT>
__global__ void k_gemm_wmma(const float* __restrict__ A, const float* __restrict__ B,
                            float* __restrict__ C, int nrows) {
  constexpr int KDIM  = 128;
  constexpr int BR    = 64;                 // A rows staged per block
  constexpr int LD    = KDIM + 8;           // padded LDS row (bf16 elements)
  constexpr int CT    = FOUT / 16;          // col tiles
  constexpr int NTILE = (BR / 16) * CT;     // 16x16 tiles per block

  __shared__ __bf16 Alds[BR * LD];          // [row][k]
  __shared__ __bf16 Blds[FOUT * LD];        // transposed: [col][k]

  const int t = threadIdx.x;
  const int rowBase0 = blockIdx.x * BR;
  const bool fullBlock = (rowBase0 + BR) <= nrows;

  // ---- stage A tile (float4 coalesced reads, bf16 convert) ----
  {
    constexpr int NF4 = BR * KDIM / 4;      // 2048 float4
    for (int f4 = t; f4 < NF4; f4 += 256) {
      const int row = (f4 * 4) / KDIM;
      const int k   = (f4 * 4) % KDIM;
      int grow = rowBase0 + row;
      if (grow >= nrows) grow = nrows - 1;  // clamp; stores guarded later
      const float4 v = ((const float4*)(A + (long long)grow * KDIM))[k >> 2];
      __bf16* dst = &Alds[row * LD + k];
      dst[0] = (__bf16)v.x; dst[1] = (__bf16)v.y;
      dst[2] = (__bf16)v.z; dst[3] = (__bf16)v.w;
    }
  }
  // ---- stage B transposed (float4 coalesced reads, scatter to LDS) ----
  {
    constexpr int NF4 = KDIM * FOUT / 4;
    for (int f4 = t; f4 < NF4; f4 += 256) {
      const int flat = f4 * 4;
      const int k = flat / FOUT;
      const int n = flat % FOUT;
      const float4 v = ((const float4*)B)[f4];
      Blds[(n + 0) * LD + k] = (__bf16)v.x;
      Blds[(n + 1) * LD + k] = (__bf16)v.y;
      Blds[(n + 2) * LD + k] = (__bf16)v.z;
      Blds[(n + 3) * LD + k] = (__bf16)v.w;
    }
  }
  __syncthreads();

  const int wave = t >> 5;
  const int lane = t & 31;
  const int lmod = lane & 15;
  const bool hi  = lane >= 16;
  const int hia  = hi ? 8 : 0;              // A chunk offset
  const int hib  = hi ? 8 : 0;              // B chunk offset

  for (int tile = wave; tile < NTILE; tile += 8) {
    const int rowTile = tile / CT;
    const int colTile = tile % CT;          // loop-invariant per wave
    const int arow = rowTile * 16 + lmod;
    const int bcol = colTile * 16 + lmod;

    // preload all fragments for the 4 k-steps
    bf8v af[8], bg[8];
    #pragma unroll
    for (int kk = 0; kk < 4; ++kk) {
      const bf8v* ap = (const bf8v*)&Alds[arow * LD + kk * 32 + hia];
      af[2 * kk]     = ap[0];               // K = base .. base+7
      af[2 * kk + 1] = ap[2];               // K = base+16 .. base+23
      const bf8v* bp = (const bf8v*)&Blds[bcol * LD + kk * 32 + hib];
      bg[2 * kk]     = bp[0];               // K = base .. base+7
      bg[2 * kk + 1] = bp[2];               // K = base+16 .. base+23
    }

    v8f acc = {};
    #pragma unroll
    for (int kk = 0; kk < 4; ++kk) {
      const v16bf av = __builtin_shufflevector(af[2 * kk], af[2 * kk + 1],
                                               0,1,2,3,4,5,6,7,8,9,10,11,12,13,14,15);
      const v16bf bv = __builtin_shufflevector(bg[2 * kk], bg[2 * kk + 1],
                                               0,1,2,3,4,5,6,7,8,9,10,11,12,13,14,15);
      acc = __builtin_amdgcn_wmma_f32_16x16x32_bf16(
          /*neg_a=*/false, av, /*neg_b=*/false, bv,
          /*c_mod=*/(short)0, acc, /*reuse_a=*/false, /*reuse_b=*/false);
    }

    const int rtop = rowBase0 + rowTile * 16 + hia;
    float* __restrict__ cp = C + (long long)rtop * FOUT + colTile * 16 + lmod;
    if (fullBlock) {
      #pragma unroll
      for (int r = 0; r < 8; ++r) cp[r * FOUT] = acc[r];      // immediate offsets
    } else {
      #pragma unroll
      for (int r = 0; r < 8; ++r)
        if (rtop + r < nrows) cp[r * FOUT] = acc[r];
    }
  }
}

// ---------------------------------------------------------------------------
// Per-node attention coefficients: as/ad[n,h] = sum_c h[n,h,c] * att[h,c]
// ---------------------------------------------------------------------------
__global__ void k_att(const float* __restrict__ h, const float* __restrict__ att_s,
                      const float* __restrict__ att_d, float* __restrict__ as_,
                      float* __restrict__ ad_, int n, int H, int C) {
  int i = blockIdx.x * blockDim.x + threadIdx.x;
  if (i >= n * H) return;
  const int node = i / H, hh = i % H;
  const float4* __restrict__ hp = (const float4*)(h + (long long)node * H * C + hh * C);
  const float4* __restrict__ sp = (const float4*)(att_s + hh * C);
  const float4* __restrict__ dp = (const float4*)(att_d + hh * C);
  float s = 0.f, d = 0.f;
  for (int c = 0; c < C / 4; ++c) {
    const float4 hv = hp[c], sv = sp[c], dv = dp[c];
    s += hv.x * sv.x + hv.y * sv.y + hv.z * sv.z + hv.w * sv.w;
    d += hv.x * dv.x + hv.y * dv.y + hv.z * dv.z + hv.w * dv.w;
  }
  as_[i] = s;
  ad_[i] = d;
}

__device__ __forceinline__ void edge_sd(const int* __restrict__ ei, int e, int E,
                                        int& s, int& d) {
  if (e < E) { s = ei[e]; d = ei[E + e]; }
  else       { s = e - E; d = e - E; }      // self loops appended
}

// segment max over dst (after leaky-relu)
__global__ void k_edge_max(const int* __restrict__ ei, const float* __restrict__ as_,
                           const float* __restrict__ ad_, float* __restrict__ m,
                           int E, int n, int H) {
  int e = blockIdx.x * blockDim.x + threadIdx.x;
  if (e >= E + n) return;
  int s, d; edge_sd(ei, e, E, s, d);
  for (int h = 0; h < H; ++h) {
    float v = lrelu(as_[s * H + h] + ad_[d * H + h]);
    atomicMaxF(&m[d * H + h], v);
  }
}

// segment sum of exp(e - m[dst])
__global__ void k_edge_sum(const int* __restrict__ ei, const float* __restrict__ as_,
                           const float* __restrict__ ad_, const float* __restrict__ m,
                           float* __restrict__ ssum, int E, int n, int H) {
  int e = blockIdx.x * blockDim.x + threadIdx.x;
  if (e >= E + n) return;
  int s, d; edge_sd(ei, e, E, s, d);
  for (int h = 0; h < H; ++h) {
    float v = lrelu(as_[s * H + h] + ad_[d * H + h]);
    atomicAdd(&ssum[d * H + h], expf(v - m[d * H + h]));
  }
}

// alpha[e,h] = exp(e - m[dst]) / (sum[dst] + 1e-16)
__global__ void k_edge_alpha(const int* __restrict__ ei, const float* __restrict__ as_,
                             const float* __restrict__ ad_, const float* __restrict__ m,
                             const float* __restrict__ ssum, float* __restrict__ alpha,
                             int E, int n, int H) {
  int e = blockIdx.x * blockDim.x + threadIdx.x;
  if (e >= E + n) return;
  int s, d; edge_sd(ei, e, E, s, d);
  for (int h = 0; h < H; ++h) {
    float v = lrelu(as_[s * H + h] + ad_[d * H + h]);
    alpha[(long long)e * H + h] = expf(v - m[d * H + h]) / (ssum[d * H + h] + 1e-16f);
  }
}

// acc[dst, c] += alpha[e, c/C] * h[src, c]   (one thread per (edge, channel))
__global__ void k_scatter(const int* __restrict__ ei, const float* __restrict__ alpha,
                          const float* __restrict__ h, float* __restrict__ acc,
                          int E, int n, int H, int C) {
  const int HC = H * C;
  long long idx = (long long)blockIdx.x * blockDim.x + threadIdx.x;
  if (idx >= (long long)(E + n) * HC) return;
  const int e = (int)(idx / HC);
  const int c = (int)(idx % HC);
  int s, d; edge_sd(ei, e, E, s, d);
  const float a = alpha[(long long)e * H + (c / C)];
  atomicAdd(&acc[(long long)d * HC + c], a * h[(long long)s * HC + c]);
}

// x = BN( [relu]( x + bias ) ), in place, 4 consecutive channels per thread
__global__ void k_post4(float4* __restrict__ x, const float* __restrict__ b,
                        const float* __restrict__ gamma, const float* __restrict__ beta,
                        const float* __restrict__ mean, const float* __restrict__ var,
                        long long n4, int F, int do_relu) {
  long long i = (long long)blockIdx.x * blockDim.x + threadIdx.x;
  if (i >= n4) return;
  const int c = (int)((i * 4) % F);
  float4 v = x[i];
  float* vp = (float*)&v;
  #pragma unroll
  for (int j = 0; j < 4; ++j) {
    float u = vp[j] + b[c + j];
    if (do_relu) u = fmaxf(u, 0.f);
    vp[j] = gamma[c + j] * (u - mean[c + j]) * rsqrtf(var[c + j] + BN_EPS) + beta[c + j];
  }
  x[i] = v;
}

// ---------------------------------------------------------------------------
extern "C" void kernel_launch(void* const* d_in, const int* in_sizes, int n_in,
                              void* d_out, int out_size, void* d_ws, size_t ws_size,
                              hipStream_t stream) {
  (void)n_in; (void)out_size; (void)ws_size;
  const float* x    = (const float*)d_in[0];
  const int*   ei   = (const int*)  d_in[1];
  const float* W1   = (const float*)d_in[2];
  const float* atS1 = (const float*)d_in[3];
  const float* atD1 = (const float*)d_in[4];
  const float* b1   = (const float*)d_in[5];
  const float* W2   = (const float*)d_in[6];
  const float* atS2 = (const float*)d_in[7];
  const float* atD2 = (const float*)d_in[8];
  const float* b2   = (const float*)d_in[9];
  const float* g1   = (const float*)d_in[10];
  const float* be1  = (const float*)d_in[11];
  const float* mn1  = (const float*)d_in[12];
  const float* vr1  = (const float*)d_in[13];
  const float* g2   = (const float*)d_in[14];
  const float* be2  = (const float*)d_in[15];
  const float* mn2  = (const float*)d_in[16];
  const float* vr2  = (const float*)d_in[17];

  const int IN_C = 128, HID = 64, HEADS = 2, OUT_C = 64;
  const int HC1 = HEADS * HID;              // 128
  const int N  = in_sizes[0] / IN_C;        // 100000
  const int E  = in_sizes[1] / 2;           // 1600000
  const int ET = E + N;                     // edges + self loops

  // workspace layout (floats)
  float* p   = (float*)d_ws;
  float* h1   = p; p += (size_t)N * HC1;
  float* acc1 = p; p += (size_t)N * HC1;
  float* h2   = p; p += (size_t)N * OUT_C;
  float* as1  = p; p += (size_t)N * HEADS;
  float* ad1  = p; p += (size_t)N * HEADS;
  float* m1   = p; p += (size_t)N * HEADS;
  float* s1   = p; p += (size_t)N * HEADS;
  float* al1  = p; p += (size_t)ET * HEADS;
  float* as2  = p; p += (size_t)N;
  float* ad2  = p; p += (size_t)N;
  float* m2   = p; p += (size_t)N;
  float* s2   = p; p += (size_t)N;
  float* al2  = p; p += (size_t)ET;
  float* outp = (float*)d_out;

  const int T = 256;
  auto cdiv = [](long long a, long long b) { return (unsigned)((a + b - 1) / b); };

  // --- init accumulators (harness poisons d_out/d_ws) ---
  k_fill<<<cdiv((long long)N * HC1, T), T, 0, stream>>>(acc1, 0.f, (long long)N * HC1);
  k_fill<<<cdiv((long long)N * HEADS, T), T, 0, stream>>>(m1, -INFINITY, (long long)N * HEADS);
  k_fill<<<cdiv((long long)N * HEADS, T), T, 0, stream>>>(s1, 0.f, (long long)N * HEADS);
  k_fill<<<cdiv((long long)N * OUT_C, T), T, 0, stream>>>(outp, 0.f, (long long)N * OUT_C);
  k_fill<<<cdiv(N, T), T, 0, stream>>>(m2, -INFINITY, N);
  k_fill<<<cdiv(N, T), T, 0, stream>>>(s2, 0.f, N);

  // --- layer 1 ---
  k_gemm_wmma<128><<<cdiv(N, 64), 256, 0, stream>>>(x, W1, h1, N);
  k_att<<<cdiv((long long)N * HEADS, T), T, 0, stream>>>(h1, atS1, atD1, as1, ad1, N, HEADS, HID);
  k_edge_max<<<cdiv(ET, T), T, 0, stream>>>(ei, as1, ad1, m1, E, N, HEADS);
  k_edge_sum<<<cdiv(ET, T), T, 0, stream>>>(ei, as1, ad1, m1, s1, E, N, HEADS);
  k_edge_alpha<<<cdiv(ET, T), T, 0, stream>>>(ei, as1, ad1, m1, s1, al1, E, N, HEADS);
  k_scatter<<<cdiv((long long)ET * HC1, T), T, 0, stream>>>(ei, al1, h1, acc1, E, N, HEADS, HID);
  k_post4<<<cdiv((long long)N * HC1 / 4, T), T, 0, stream>>>((float4*)acc1, b1, g1, be1, mn1, vr1,
                                                             (long long)N * HC1 / 4, HC1, 1);

  // --- layer 2 ---
  k_gemm_wmma<64><<<cdiv(N, 64), 256, 0, stream>>>(acc1, W2, h2, N);
  k_att<<<cdiv(N, T), T, 0, stream>>>(h2, atS2, atD2, as2, ad2, N, 1, OUT_C);
  k_edge_max<<<cdiv(ET, T), T, 0, stream>>>(ei, as2, ad2, m2, E, N, 1);
  k_edge_sum<<<cdiv(ET, T), T, 0, stream>>>(ei, as2, ad2, m2, s2, E, N, 1);
  k_edge_alpha<<<cdiv(ET, T), T, 0, stream>>>(ei, as2, ad2, m2, s2, al2, E, N, 1);
  k_scatter<<<cdiv((long long)ET * OUT_C, T), T, 0, stream>>>(ei, al2, h2, outp, E, N, 1, OUT_C);
  k_post4<<<cdiv((long long)N * OUT_C / 4, T), T, 0, stream>>>((float4*)outp, b2, g2, be2, mn2, vr2,
                                                               (long long)N * OUT_C / 4, OUT_C, 0);
}
